// DetectionClassificationModel_80315888435791
// MI455X (gfx1250) — matrix operand order repaired and checked by reading, not compiled, not run on hardware
//
#include <hip/hip_runtime.h>
#include <hip/hip_fp16.h>
#include <math.h>

// ---------------------------------------------------------------------------
// MI455X (gfx1250) implementation.
// ~18 GFLOP total, ~13MB input -> trivially under the 23.3 TB/s roofline, so
// the design goal is: one workgroup per sample, activations LDS-resident,
// every GEMM on v_wmma_f32_16x16x32_f16 (f16 operands, f32 accumulate),
// wave32 shuffle reductions for layernorm/softmax, and TDM tensor_load_to_lds
// to stage the per-sample feature tile (async-tensor path, TENSORcnt).
// Weights are read straight from global (hot in the 192MB L2 across blocks).
// ---------------------------------------------------------------------------

typedef __attribute__((ext_vector_type(16))) _Float16 v16h;
typedef __attribute__((ext_vector_type(8)))  float    v8f;
typedef __attribute__((ext_vector_type(4)))  unsigned int u32x4;
typedef __attribute__((ext_vector_type(8)))  int      i32x8;
typedef __attribute__((ext_vector_type(4)))  int      i32x4;

__device__ __forceinline__ float gelu_exact(float x) {
  return 0.5f * x * (1.0f + erff(x * 0.70710678118654752f));
}

// ---- TDM: DMA a contiguous run of f32 (1 x nelem tile) global -> LDS ------
// D# layout per CDNA5 ISA 8.3/8.4. Uniform (scalar) operands -> SGPR groups.
__device__ __forceinline__ void tdm_load_f32(unsigned lds_off, const float* gptr,
                                             unsigned nelem) {
  unsigned long long ga = (unsigned long long)(uintptr_t)gptr;
  u32x4 g0;
  g0[0] = 1u;                                   // count=1 (user descriptor)
  g0[1] = lds_off;                              // lds_addr (bytes)
  g0[2] = (unsigned)(ga & 0xFFFFFFFFu);         // global_addr[31:0]
  g0[3] = (unsigned)((ga >> 32) & 0x01FFFFFFu)  // global_addr[56:32]
          | (2u << 30);                         // type=2 ("image")
  i32x8 g1;
  g1[0] = (int)(2u << 16);                      // data_size=2 -> 4 bytes
  g1[1] = (int)((nelem & 0xFFFFu) << 16);       // tensor_dim0[15:0]
  g1[2] = (int)(((nelem >> 16) & 0xFFFFu)       // tensor_dim0[31:16]
          | (1u << 16));                        // tensor_dim1 = 1
  g1[3] = (int)((nelem & 0xFFFFu) << 16);       // tile_dim0 = nelem
  g1[4] = 1;                                    // tile_dim1 = 1
  g1[5] = (int)nelem;                           // tensor_dim0_stride
  g1[6] = (int)((nelem & 0xFFFFu) << 16);       // tensor_dim1_stride lo
  g1[7] = 0;
  i32x4 z4 = {};
#if defined(__clang_major__) && (__clang_major__ >= 23)
  i32x8 z8 = {};
  __builtin_amdgcn_tensor_load_to_lds(g0, g1, z4, z4, z8, 0);
#else
  __builtin_amdgcn_tensor_load_to_lds(g0, g1, z4, z4, 0);
#endif
}

__device__ __forceinline__ unsigned lds_offset_of(const void* p) {
  // LDS aperture: flat address truncates to addr[31:0] (ISA 10.2).
  return (unsigned)(uintptr_t)p;
}

// ---- WMMA fragment builders (wave32; A/B use contiguous k order, C/D per ISA 7.12.2) --------
// A 16x32 f16: lane L holds row M=L%16; lanes 0-15 K=0..15, lanes 16-31 K=16..31.
template <typename F>
__device__ __forceinline__ v16h fragA(int lane, F get) {
  v16h a;
  const int m  = lane & 15;
  const int kb = (lane & 16) ? 16 : 0;
#pragma unroll
  for (int e = 0; e < 16; ++e) {
    int k = kb + e;
    a[e] = get(m, k);
  }
  return a;
}
// B 32x16 f16: lane L holds col N=L%16; lanes 0-15 K=0..15, lanes 16-31 K=16..31.
template <typename F>
__device__ __forceinline__ v16h fragB(int lane, F get) {
  v16h b;
  const int n  = lane & 15;
  const int kb = (lane & 16) ? 16 : 0;
#pragma unroll
  for (int e = 0; e < 16; ++e) b[e] = get(kb + e, n);
  return b;
}
// C/D 16x16 f32: elem r -> row M = r + 8*(lane>=16), col N = lane%16.
template <typename F>
__device__ __forceinline__ void storeC(int lane, v8f c, F put) {
  const int n  = lane & 15;
  const int mb = (lane & 16) ? 8 : 0;
#pragma unroll
  for (int r = 0; r < 8; ++r) put(mb + r, n, c[r]);
}

__device__ __forceinline__ v8f wmma16(v16h a, v16h b, v8f c) {
  return __builtin_amdgcn_wmma_f32_16x16x32_f16(false, a, false, b, (short)0, c,
                                                false, false);
}

// ---- wave32 reductions ----------------------------------------------------
__device__ __forceinline__ float wsum32(float v) {
  v += __shfl_xor(v, 16, 32);
  v += __shfl_xor(v, 8, 32);
  v += __shfl_xor(v, 4, 32);
  v += __shfl_xor(v, 2, 32);
  v += __shfl_xor(v, 1, 32);
  return v;
}
__device__ __forceinline__ float wsum_half16(float v) {
  v += __shfl_xor(v, 8, 32);
  v += __shfl_xor(v, 4, 32);
  v += __shfl_xor(v, 2, 32);
  v += __shfl_xor(v, 1, 32);
  return v;
}

// LayerNorm over 32 channels for 16 rows; one lane per channel (wave32 fit).
__device__ __forceinline__ void ln_rows32(float* xf, int rbase, const float* w,
                                          const float* bb, int lane) {
  for (int r = 0; r < 16; ++r) {
    float* row = xf + (rbase + r) * 32;
    float v    = row[lane];
    float mean = wsum32(v) * (1.f / 32.f);
    float d    = v - mean;
    float var  = wsum32(d * d) * (1.f / 32.f);
    float rs   = rsqrtf(var + 1e-5f);
    row[lane]  = d * rs * w[lane] + bb[lane];
  }
}

// LayerNorm over 16 channels, two rows per pass (half-wave reductions).
__device__ __forceinline__ void ln_rows16(float* xf, int rbase, const float* w,
                                          const float* bb, int lane) {
  const int c  = lane & 15;
  const int rh = lane >> 4;
  for (int r = 0; r < 16; r += 2) {
    float* row = xf + (rbase + r + rh) * 16;
    float v    = row[c];
    float mean = wsum_half16(v) * (1.f / 16.f);
    float d    = v - mean;
    float var  = wsum_half16(d * d) * (1.f / 16.f);
    float rs   = rsqrtf(var + 1e-5f);
    row[c]     = d * rs * w[c] + bb[c];
  }
}

// ---- parameter bundles ----------------------------------------------------
struct EncLayer {
  const float *W1, *W2, *Wo, *Wqkv, *b1, *b2, *bo, *bqkv;
  const float *ln1b, *ln1w, *ln2b, *ln2w;
};

struct ObjP {
  const float* feats;  // [B,128,7]
  const float* emb;    // [22,16]
  const float *Wnum, *bnum;  // [16,6],[16]
  const float *onw, *onb;    // [6]
  const float *lnw, *lnb;    // obj_ln [32]
  EncLayer L[2];
  const float *fnw, *fnb;    // obj_norm [32]
  const float *Wop, *bop;    // [16,32],[16]
  float* out;                // [B,16]
};

struct FigP {
  const float* feats;  // [B,32,20]
  const float *Wcc, *bcc;    // [16,20],[16]
  const float *lnw, *lnb;    // fig_ln [16]
  EncLayer L[2];
  const float *fnw, *fnb;    // fig_norm [16]
  float* out;                // [B,16]
  int nB;
};

struct HeadP {
  const float *obj, *fig, *grp;
  const float *W0, *b0, *W1, *b1, *W2, *b2;
  float* out;
  int nB;
};

// ===========================================================================
// Object branch: seq=128, d=32, 4 heads (dh=8), FFN=64, 2 layers.
// One block per sample, 2 waves (64 threads), each wave owns 4 M-tiles.
// ===========================================================================
__global__ __launch_bounds__(64) void obj_encoder_kernel(ObjP P) {
  __shared__ float    lfeat[128 * 7];     // TDM-staged features  3.5KB
  __shared__ float    xf[128 * 32];       // activations (f32)   16KB
  __shared__ _Float16 qh[128 * 32];       // q (pre-scaled)       8KB
  __shared__ _Float16 kh[128 * 32];       //                      8KB
  __shared__ _Float16 vh[128 * 32];       //                      8KB
  __shared__ _Float16 oh[128 * 32];       // attn out (f16)       8KB
  __shared__ _Float16 sc[2 * 16 * 128];   // per-wave scores/FFN  8KB
  __shared__ float    maskf[128];
  __shared__ float    pooled[32];

  const int tid  = threadIdx.x;
  const int lane = tid & 31;
  const int wave = tid >> 5;
  const int b    = blockIdx.x;
  _Float16* scw  = sc + wave * (16 * 128);

  // ---- TDM DMA: this sample's feature tile global -> LDS ------------------
  if (wave == 0) {
    tdm_load_f32(lds_offset_of(lfeat), P.feats + (size_t)b * 128 * 7, 128 * 7);
    __builtin_amdgcn_s_wait_tensorcnt(0);
  }
  __syncthreads();

  // ---- embedding: label emb + LN(num)*Wnum proj + PE + obj_ln -------------
  const float LOG1E4_D = logf(10000.f) / 32.f;
  for (int p = tid; p < 128; p += 64) {
    const float* f = lfeat + p * 7;
    int label = (int)f[0];
    maskf[p]  = (label != 21 && label != 20) ? 1.f : 0.f;
    float m = 0.f;
#pragma unroll
    for (int j = 0; j < 6; ++j) m += f[1 + j];
    m *= (1.f / 6.f);
    float v = 0.f;
#pragma unroll
    for (int j = 0; j < 6; ++j) { float d = f[1 + j] - m; v += d * d; }
    float rsn = rsqrtf(v * (1.f / 6.f) + 1e-5f);
    float nf[6];
#pragma unroll
    for (int j = 0; j < 6; ++j) nf[j] = (f[1 + j] - m) * rsn * P.onw[j] + P.onb[j];

    float row[32];
#pragma unroll
    for (int c = 0; c < 32; ++c) {
      float val;
      if (c < 16) {
        val = P.emb[label * 16 + c];
      } else {
        val = P.bnum[c - 16];
#pragma unroll
        for (int j = 0; j < 6; ++j) val += P.Wnum[(c - 16) * 6 + j] * nf[j];
      }
      float ang = (float)p * expf(-(float)(c & ~1) * LOG1E4_D);
      val += (c & 1) ? cosf(ang) : sinf(ang);
      row[c] = val;
    }
    // obj_ln (serial per position, 32 ch in regs)
    float mu = 0.f;
#pragma unroll
    for (int c = 0; c < 32; ++c) mu += row[c];
    mu *= (1.f / 32.f);
    float var = 0.f;
#pragma unroll
    for (int c = 0; c < 32; ++c) { float d = row[c] - mu; var += d * d; }
    float rs = rsqrtf(var * (1.f / 32.f) + 1e-5f);
#pragma unroll
    for (int c = 0; c < 32; ++c)
      xf[p * 32 + c] = (row[c] - mu) * rs * P.lnw[c] + P.lnb[c];
  }
  __syncthreads();

  // ---- encoder layers -----------------------------------------------------
  for (int lay = 0; lay < 2; ++lay) {
    const EncLayer Lp = P.L[lay];

    // Prefetch next layer's big weights into L2 while layer 0 computes.
    if (lay == 0) {
      __builtin_prefetch(P.L[1].Wqkv + tid * 48, 0, 1);
      __builtin_prefetch(P.L[1].W1 + tid * 32, 0, 1);
      __builtin_prefetch(P.L[1].W2 + tid * 32, 0, 1);
    }

    // QKV: x[128x32] @ Wqkv^T[32x96]  (6 N-tiles, K=32 exact)
    for (int mt = wave; mt < 8; mt += 2) {
      const int R = mt * 16;
      v16h a = fragA(lane, [&](int m, int k) -> _Float16 {
        return (_Float16)xf[(R + m) * 32 + k];
      });
#pragma unroll
      for (int nt = 0; nt < 6; ++nt) {
        v16h bv = fragB(lane, [&](int k, int n) -> _Float16 {
          return (_Float16)Lp.Wqkv[(nt * 16 + n) * 32 + k];
        });
        v8f acc = {};
        acc = wmma16(a, bv, acc);
        storeC(lane, acc, [&](int m, int n, float val) {
          int   c    = nt * 16 + n;
          float full = val + Lp.bqkv[c];
          if (c < 32)
            qh[(R + m) * 32 + c] = (_Float16)(full * 0.35355339059f);  // /sqrt(8)
          else if (c < 64)
            kh[(R + m) * 32 + (c - 32)] = (_Float16)full;
          else
            vh[(R + m) * 32 + (c - 64)] = (_Float16)full;
        });
      }
    }
    __syncthreads();

    // attention + out-proj + FFN, per M-tile (wave-local past this point)
    for (int mt = wave; mt < 8; mt += 2) {
      const int R = mt * 16;

      for (int h = 0; h < 4; ++h) {
        // scores: channel-masked f16 WMMA (8 live K channels of 32)
        v16h aq = fragA(lane, [&](int m, int k) -> _Float16 {
          return ((k >> 3) == h) ? qh[(R + m) * 32 + k] : (_Float16)0.f;
        });
#pragma unroll
        for (int nt = 0; nt < 8; ++nt) {
          v16h bk = fragB(lane, [&](int k, int n) -> _Float16 {
            return ((k >> 3) == h) ? kh[(nt * 16 + n) * 32 + k] : (_Float16)0.f;
          });
          v8f acc = {};
          acc = wmma16(aq, bk, acc);
          storeC(lane, acc, [&](int m, int n, float val) {
            int key = nt * 16 + n;
            scw[m * 128 + key] =
                (_Float16)((maskf[key] != 0.f) ? val : -30000.f);
          });
        }
        // softmax over 128 keys (2 lanes per row, combine via shfl_xor 16)
        {
          int   r = lane & 15, half = (lane >> 4) * 64;
          float mx = -1e30f;
#pragma unroll
          for (int kk = 0; kk < 64; ++kk)
            mx = fmaxf(mx, (float)scw[r * 128 + half + kk]);
          mx = fmaxf(mx, __shfl_xor(mx, 16, 32));
          float sum = 0.f;
#pragma unroll
          for (int kk = 0; kk < 64; ++kk) {
            float e = expf((float)scw[r * 128 + half + kk] - mx);
            scw[r * 128 + half + kk] = (_Float16)e;
            sum += e;
          }
          sum += __shfl_xor(sum, 16, 32);
          float inv = 1.f / sum;
#pragma unroll
          for (int kk = 0; kk < 64; ++kk)
            scw[r * 128 + half + kk] =
                (_Float16)((float)scw[r * 128 + half + kk] * inv);
        }
        // attn x V: K=128 -> 4 chained WMMAs, 8 valid out cols
        v8f acc = {};
#pragma unroll
        for (int kc = 0; kc < 4; ++kc) {
          v16h ap = fragA(lane, [&](int m, int k) -> _Float16 {
            return scw[m * 128 + kc * 32 + k];
          });
          v16h bvv = fragB(lane, [&](int k, int n) -> _Float16 {
            return (n < 8) ? vh[(kc * 32 + k) * 32 + h * 8 + n] : (_Float16)0.f;
          });
          acc = wmma16(ap, bvv, acc);
        }
        storeC(lane, acc, [&](int m, int n, float val) {
          if (n < 8) oh[(R + m) * 32 + h * 8 + n] = (_Float16)val;
        });
      }

      // out-proj + residual + LN1
      {
        v16h ao = fragA(lane, [&](int m, int k) -> _Float16 {
          return oh[(R + m) * 32 + k];
        });
#pragma unroll
        for (int nt = 0; nt < 2; ++nt) {
          v16h bw = fragB(lane, [&](int k, int n) -> _Float16 {
            return (_Float16)Lp.Wo[(nt * 16 + n) * 32 + k];
          });
          v8f acc = {};
          acc = wmma16(ao, bw, acc);
          storeC(lane, acc, [&](int m, int n, float val) {
            int c = nt * 16 + n;
            xf[(R + m) * 32 + c] += val + Lp.bo[c];
          });
        }
        ln_rows32(xf, R, Lp.ln1w, Lp.ln1b, lane);
      }

      // FFN (hidden reuses the per-wave score buffer)
      {
        _Float16* hw = scw;  // [16 x 64] f16
        v16h ax = fragA(lane, [&](int m, int k) -> _Float16 {
          return (_Float16)xf[(R + m) * 32 + k];
        });
#pragma unroll
        for (int nt = 0; nt < 4; ++nt) {
          v16h bw = fragB(lane, [&](int k, int n) -> _Float16 {
            return (_Float16)Lp.W1[(nt * 16 + n) * 32 + k];
          });
          v8f acc = {};
          acc = wmma16(ax, bw, acc);
          storeC(lane, acc, [&](int m, int n, float val) {
            int c = nt * 16 + n;
            hw[m * 64 + c] = (_Float16)gelu_exact(val + Lp.b1[c]);
          });
        }
#pragma unroll
        for (int nt = 0; nt < 2; ++nt) {
          v8f acc = {};
#pragma unroll
          for (int kc = 0; kc < 2; ++kc) {
            v16h ah = fragA(lane, [&](int m, int k) -> _Float16 {
              return hw[m * 64 + kc * 32 + k];
            });
            v16h bw = fragB(lane, [&](int k, int n) -> _Float16 {
              return (_Float16)Lp.W2[(nt * 16 + n) * 64 + kc * 32 + k];
            });
            acc = wmma16(ah, bw, acc);
          }
          storeC(lane, acc, [&](int m, int n, float val) {
            int c = nt * 16 + n;
            xf[(R + m) * 32 + c] += val + Lp.b2[c];
          });
        }
        ln_rows32(xf, R, Lp.ln2w, Lp.ln2b, lane);
      }
    }
    __syncthreads();
  }

  // ---- final norm, masked mean pool, Wop projection -----------------------
  for (int mt = wave; mt < 8; mt += 2) ln_rows32(xf, mt * 16, P.fnw, P.fnb, lane);
  __syncthreads();
  if (wave == 0) {
    float s = 0.f, cnt = 0.f;
    for (int p = 0; p < 128; ++p) {
      s += xf[p * 32 + lane] * maskf[p];
      cnt += maskf[p];
    }
    pooled[lane] = s / cnt;
  }
  if (wave == 0 && lane < 16) {
    float o = P.bop[lane];
#pragma unroll
    for (int c = 0; c < 32; ++c) o += pooled[c] * P.Wop[lane * 32 + c];
    P.out[(size_t)b * 16 + lane] = o;
  }
}

// ===========================================================================
// Figure branch: seq=32, d=16, 4 heads (dh=4), FFN=64, 2 layers.
// One wave per sample, 4 samples per block; fully wave-local (no barriers:
// per-wave TDM DMA + per-wave TENSORcnt wait).
// ===========================================================================
__global__ __launch_bounds__(128) void fig_encoder_kernel(FigP P) {
  __shared__ float    lfeats[4][32 * 20];
  __shared__ float    xfs[4][32 * 16];
  __shared__ _Float16 qhs[4][32 * 16];
  __shared__ _Float16 khs[4][32 * 16];
  __shared__ _Float16 vhs[4][32 * 16];
  __shared__ _Float16 ohs[4][32 * 16];
  __shared__ _Float16 scs[4][16 * 64];
  __shared__ float    maskfs[4][32];

  const int lane = threadIdx.x & 31;
  const int wave = threadIdx.x >> 5;
  const int b    = blockIdx.x * 4 + wave;
  if (b >= P.nB) return;  // uniform per wave

  float*    lfeat = lfeats[wave];
  float*    xf    = xfs[wave];
  _Float16* qh    = qhs[wave];
  _Float16* kh    = khs[wave];
  _Float16* vh    = vhs[wave];
  _Float16* oh    = ohs[wave];
  _Float16* sc    = scs[wave];
  float*    maskf = maskfs[wave];

  // ---- per-wave TDM DMA of this sample's features -------------------------
  tdm_load_f32(lds_offset_of(lfeat), P.feats + (size_t)b * 32 * 20, 32 * 20);
  __builtin_amdgcn_s_wait_tensorcnt(0);

  // ---- input proj + PE + fig_ln (one lane per position) -------------------
  {
    const int    p = lane;
    const float* f = lfeat + p * 20;
    int label = (int)f[0];
    maskf[p]  = (label != 21 && label != 20) ? 1.f : 0.f;
    const float LOG1E4_D = logf(10000.f) / 16.f;
    float row[16];
#pragma unroll
    for (int c = 0; c < 16; ++c) {
      float val = P.bcc[c];
#pragma unroll
      for (int j = 0; j < 20; ++j) val += P.Wcc[c * 20 + j] * f[j];
      float ang = (float)p * expf(-(float)(c & ~1) * LOG1E4_D);
      val += (c & 1) ? cosf(ang) : sinf(ang);
      row[c] = val;
    }
    float mu = 0.f;
#pragma unroll
    for (int c = 0; c < 16; ++c) mu += row[c];
    mu *= (1.f / 16.f);
    float var = 0.f;
#pragma unroll
    for (int c = 0; c < 16; ++c) { float d = row[c] - mu; var += d * d; }
    float rs = rsqrtf(var * (1.f / 16.f) + 1e-5f);
#pragma unroll
    for (int c = 0; c < 16; ++c)
      xf[p * 16 + c] = (row[c] - mu) * rs * P.lnw[c] + P.lnb[c];
  }

  for (int lay = 0; lay < 2; ++lay) {
    const EncLayer Lp = P.L[lay];

    // QKV for both M-tiles (K=16 zero-padded to 32)
    for (int mt = 0; mt < 2; ++mt) {
      const int R = mt * 16;
      v16h a = fragA(lane, [&](int m, int k) -> _Float16 {
        return (k < 16) ? (_Float16)xf[(R + m) * 16 + k] : (_Float16)0.f;
      });
#pragma unroll
      for (int nt = 0; nt < 3; ++nt) {
        v16h bv = fragB(lane, [&](int k, int n) -> _Float16 {
          return (k < 16) ? (_Float16)Lp.Wqkv[(nt * 16 + n) * 16 + k]
                          : (_Float16)0.f;
        });
        v8f acc = {};
        acc = wmma16(a, bv, acc);
        storeC(lane, acc, [&](int m, int n, float val) {
          int   c    = nt * 16 + n;
          float full = val + Lp.bqkv[c];
          if (c < 16)
            qh[(R + m) * 16 + c] = (_Float16)(full * 0.5f);  // /sqrt(4)
          else if (c < 32)
            kh[(R + m) * 16 + (c - 16)] = (_Float16)full;
          else
            vh[(R + m) * 16 + (c - 32)] = (_Float16)full;
        });
      }
    }

    for (int mt = 0; mt < 2; ++mt) {
      const int R = mt * 16;
      for (int h = 0; h < 4; ++h) {
        v16h aq = fragA(lane, [&](int m, int k) -> _Float16 {
          return (k < 16 && (k >> 2) == h) ? qh[(R + m) * 16 + k]
                                           : (_Float16)0.f;
        });
#pragma unroll
        for (int nt = 0; nt < 2; ++nt) {
          v16h bk = fragB(lane, [&](int k, int n) -> _Float16 {
            return (k < 16 && (k >> 2) == h) ? kh[(nt * 16 + n) * 16 + k]
                                             : (_Float16)0.f;
          });
          v8f acc = {};
          acc = wmma16(aq, bk, acc);
          storeC(lane, acc, [&](int m, int n, float val) {
            int key = nt * 16 + n;
            sc[m * 32 + key] =
                (_Float16)((maskf[key] != 0.f) ? val : -30000.f);
          });
        }
        // softmax over 32 keys (2 lanes per row)
        {
          int   r = lane & 15, hb = (lane >> 4) * 16;
          float mx = -1e30f;
#pragma unroll
          for (int kk = 0; kk < 16; ++kk)
            mx = fmaxf(mx, (float)sc[r * 32 + hb + kk]);
          mx = fmaxf(mx, __shfl_xor(mx, 16, 32));
          float sum = 0.f;
#pragma unroll
          for (int kk = 0; kk < 16; ++kk) {
            float e = expf((float)sc[r * 32 + hb + kk] - mx);
            sc[r * 32 + hb + kk] = (_Float16)e;
            sum += e;
          }
          sum += __shfl_xor(sum, 16, 32);
          float inv = 1.f / sum;
#pragma unroll
          for (int kk = 0; kk < 16; ++kk)
            sc[r * 32 + hb + kk] =
                (_Float16)((float)sc[r * 32 + hb + kk] * inv);
        }
        // attn x V: K=32 exact, 4 valid out cols
        v16h ap = fragA(lane, [&](int m, int k) -> _Float16 {
          return sc[m * 32 + k];
        });
        v16h bvv = fragB(lane, [&](int k, int n) -> _Float16 {
          return (n < 4) ? vh[k * 16 + h * 4 + n] : (_Float16)0.f;
        });
        v8f acc = {};
        acc = wmma16(ap, bvv, acc);
        storeC(lane, acc, [&](int m, int n, float val) {
          if (n < 4) oh[(R + m) * 16 + h * 4 + n] = (_Float16)val;
        });
      }

      // out-proj + residual + LN1
      {
        v16h ao = fragA(lane, [&](int m, int k) -> _Float16 {
          return (k < 16) ? oh[(R + m) * 16 + k] : (_Float16)0.f;
        });
        v16h bw = fragB(lane, [&](int k, int n) -> _Float16 {
          return (k < 16) ? (_Float16)Lp.Wo[n * 16 + k] : (_Float16)0.f;
        });
        v8f acc = {};
        acc = wmma16(ao, bw, acc);
        storeC(lane, acc, [&](int m, int n, float val) {
          xf[(R + m) * 16 + n] += val + Lp.bo[n];
        });
        ln_rows16(xf, R, Lp.ln1w, Lp.ln1b, lane);
      }

      // FFN
      {
        _Float16* hw = sc;  // [16 x 64]
        v16h ax = fragA(lane, [&](int m, int k) -> _Float16 {
          return (k < 16) ? (_Float16)xf[(R + m) * 16 + k] : (_Float16)0.f;
        });
#pragma unroll
        for (int nt = 0; nt < 4; ++nt) {
          v16h b1w = fragB(lane, [&](int k, int n) -> _Float16 {
            return (k < 16) ? (_Float16)Lp.W1[(nt * 16 + n) * 16 + k]
                            : (_Float16)0.f;
          });
          v8f acc = {};
          acc = wmma16(ax, b1w, acc);
          storeC(lane, acc, [&](int m, int n, float val) {
            int c = nt * 16 + n;
            hw[m * 64 + c] = (_Float16)gelu_exact(val + Lp.b1[c]);
          });
        }
        v8f acc = {};
#pragma unroll
        for (int kc = 0; kc < 2; ++kc) {
          v16h ah = fragA(lane, [&](int m, int k) -> _Float16 {
            return hw[m * 64 + kc * 32 + k];
          });
          v16h b2w = fragB(lane, [&](int k, int n) -> _Float16 {
            return (_Float16)Lp.W2[n * 64 + kc * 32 + k];
          });
          acc = wmma16(ah, b2w, acc);
        }
        storeC(lane, acc, [&](int m, int n, float val) {
          xf[(R + m) * 16 + n] += val + Lp.b2[n];
        });
        ln_rows16(xf, R, Lp.ln2w, Lp.ln2b, lane);
      }
    }
  }

  // ---- final norm + masked mean pool --------------------------------------
  for (int mt = 0; mt < 2; ++mt) ln_rows16(xf, mt * 16, P.fnw, P.fnb, lane);
  if (lane < 16) {
    float s = 0.f, cnt = 0.f;
    for (int p = 0; p < 32; ++p) {
      s += xf[p * 16 + lane] * maskf[p];
      cnt += maskf[p];
    }
    P.out[(size_t)b * 16 + lane] = s / cnt;
  }
}

// ===========================================================================
// MLP head: [obj16 | fig16 | group21] -> 16 -> 8 -> 1 (gelu, gelu, linear)
// ===========================================================================
__global__ __launch_bounds__(256) void head_kernel(HeadP P) {
  int i = blockIdx.x * 256 + threadIdx.x;
  if (i >= P.nB) return;
  float hin[53];
#pragma unroll
  for (int j = 0; j < 16; ++j) hin[j] = P.obj[(size_t)i * 16 + j];
#pragma unroll
  for (int j = 0; j < 16; ++j) hin[16 + j] = P.fig[(size_t)i * 16 + j];
#pragma unroll
  for (int j = 0; j < 21; ++j) hin[32 + j] = P.grp[(size_t)i * 21 + j];
  float h1[16];
#pragma unroll
  for (int j = 0; j < 16; ++j) {
    float s = P.b0[j];
    for (int k = 0; k < 53; ++k) s += P.W0[j * 53 + k] * hin[k];
    h1[j] = gelu_exact(s);
  }
  float h2[8];
#pragma unroll
  for (int j = 0; j < 8; ++j) {
    float s = P.b1[j];
    for (int k = 0; k < 16; ++k) s += P.W1[j * 16 + k] * h1[k];
    h2[j] = gelu_exact(s);
  }
  float s = P.b2[0];
#pragma unroll
  for (int k = 0; k < 8; ++k) s += P.W2[k] * h2[k];
  P.out[i] = s;
}

// ===========================================================================
// Host side. Input order: 5 data arrays then params flattened in jax-tree
// (alphabetical dict key) order.
// ===========================================================================
static EncLayer make_layer(void* const* d_in, int base) {
  EncLayer L;
  L.W1   = (const float*)d_in[base + 0];
  L.W2   = (const float*)d_in[base + 1];
  L.Wo   = (const float*)d_in[base + 2];
  L.Wqkv = (const float*)d_in[base + 3];
  L.b1   = (const float*)d_in[base + 4];
  L.b2   = (const float*)d_in[base + 5];
  L.bo   = (const float*)d_in[base + 6];
  L.bqkv = (const float*)d_in[base + 7];
  L.ln1b = (const float*)d_in[base + 8];
  L.ln1w = (const float*)d_in[base + 9];
  L.ln2b = (const float*)d_in[base + 10];
  L.ln2w = (const float*)d_in[base + 11];
  return L;
}

extern "C" void kernel_launch(void* const* d_in, const int* in_sizes, int n_in,
                              void* d_out, int out_size, void* d_ws,
                              size_t ws_size, hipStream_t stream) {
  (void)n_in; (void)ws_size;
  const int B = in_sizes[0] / (128 * 7);

  // params flattened alphabetically after the 5 data inputs:
  // 5:Wcc 6:Wnum 7:Wop 8:bcc 9:bnum 10:bop 11:emb_table
  // 12..17: fc0.W fc0.b fc1.W fc1.b fc2.W fc2.b
  // 18..41: fig_layers[0..1] (W1,W2,Wo,Wqkv,b1,b2,bo,bqkv,ln1_b,ln1_w,ln2_b,ln2_w)
  // 42:fig_ln_b 43:fig_ln_w 44:fig_norm.b 45:fig_norm.w
  // 46..69: obj_layers[0..1]  70:obj_ln_b 71:obj_ln_w 72:obj_norm.b 73:obj_norm.w
  // 74:onorm_b 75:onorm_w
  float* obj_ws = (float*)d_ws;            // [B,16]
  float* fig_ws = obj_ws + (size_t)B * 16; // [B,16]

  ObjP OP;
  OP.feats = (const float*)d_in[0];
  OP.emb   = (const float*)d_in[11];
  OP.Wnum  = (const float*)d_in[6];
  OP.bnum  = (const float*)d_in[9];
  OP.onw   = (const float*)d_in[75];
  OP.onb   = (const float*)d_in[74];
  OP.lnw   = (const float*)d_in[71];
  OP.lnb   = (const float*)d_in[70];
  OP.L[0]  = make_layer(d_in, 46);
  OP.L[1]  = make_layer(d_in, 58);
  OP.fnw   = (const float*)d_in[73];
  OP.fnb   = (const float*)d_in[72];
  OP.Wop   = (const float*)d_in[7];
  OP.bop   = (const float*)d_in[10];
  OP.out   = obj_ws;

  FigP FP;
  FP.feats = (const float*)d_in[1];
  FP.Wcc   = (const float*)d_in[5];
  FP.bcc   = (const float*)d_in[8];
  FP.lnw   = (const float*)d_in[43];
  FP.lnb   = (const float*)d_in[42];
  FP.L[0]  = make_layer(d_in, 18);
  FP.L[1]  = make_layer(d_in, 30);
  FP.fnw   = (const float*)d_in[45];
  FP.fnb   = (const float*)d_in[44];
  FP.out   = fig_ws;
  FP.nB    = B;

  HeadP HP;
  HP.obj = obj_ws;
  HP.fig = fig_ws;
  HP.grp = (const float*)d_in[2];
  HP.W0  = (const float*)d_in[12];
  HP.b0  = (const float*)d_in[13];
  HP.W1  = (const float*)d_in[14];
  HP.b1  = (const float*)d_in[15];
  HP.W2  = (const float*)d_in[16];
  HP.b2  = (const float*)d_in[17];
  HP.out = (float*)d_out;
  HP.nB  = (out_size > 0) ? out_size : B;

  obj_encoder_kernel<<<B, 64, 0, stream>>>(OP);
  fig_encoder_kernel<<<(B + 3) / 4, 128, 0, stream>>>(FP);
  head_kernel<<<(HP.nB + 255) / 256, 256, 0, stream>>>(HP);
}
